// HART_52467320487967
// MI455X (gfx1250) — matrix operand order repaired and checked by reading, not compiled, not run on hardware
//
// HART hypergraph-transformer for MI455X (gfx1250, wave32).
// All dense GEMMs run on v_wmma_f32_16x16x32_bf16 (bf16 in, fp32 acc),
// software-pipelined: next K-step fragments are loaded while current
// fragments feed 8 WMMAs (32x64 C tile per wave, 128x128 per block).
// Attention (~1% of FLOPs) + LayerNorm stay fp32 VALU/LDS.
#include <hip/hip_runtime.h>
#include <hip/hip_bf16.h>
#include <cstddef>
#include <cstdint>

#define D_    256
#define FF_   1024
#define NH    4
#define DH    64
#define NE    32768
#define NN    65536
#define SV    11          // V2E sequence length (CLS + 10)
#define SE    17          // E2V sequence length (CLS + 16)
#define CH    2048        // sequences per chunk
#define NTOT  100000
#define KHOP  2
#define BPRED 256

typedef __bf16 bf16;
typedef __attribute__((ext_vector_type(16))) __bf16 v16bf;
typedef __attribute__((ext_vector_type(8)))  __bf16 v8bf;
typedef __attribute__((ext_vector_type(8)))  float  v8f;

// ---------------------------------------------------------------- utilities
__global__ void f2bf_kernel(const float* __restrict__ s, bf16* __restrict__ d, int n) {
  int i = blockIdx.x * blockDim.x + threadIdx.x;
  if (i < n) d[i] = (bf16)s[i];
}

// ------------------------------------------------------------- WMMA GEMM
// C[M,N] = act(A[M,K] * W[N,K]^T + bias), A/W bf16 row-major, C bf16.
// 8 waves/block; wave -> 32x64 tile (2 M-frags x 4 N-frags, 8 WMMAs/K-step);
// block -> 128x128. K-loop is double-buffered in registers.
template<bool RELU>
__global__ __launch_bounds__(256)
void gemm_bf16_wmma(const bf16* __restrict__ A, const bf16* __restrict__ W,
                    const float* __restrict__ bias, bf16* __restrict__ Cout,
                    int M, int N, int K) {
  const int lane = threadIdx.x & 31;
  const int wave = threadIdx.x >> 5;
  const int r    = lane & 15;
  const int hi   = lane >> 4;
  const int m0   = blockIdx.y * 128 + (wave & 3) * 32;
  const int n0   = blockIdx.x * 128 + (wave >> 2) * 64;
  if (m0 >= M) return;

  const bf16* arow0 = A + (size_t)(m0 + r) * K;
  const bf16* arow1 = A + (size_t)(m0 + 16 + r) * K;
  const bf16* wrow0 = W + (size_t)(n0 + r) * K;
  const bf16* wrow1 = W + (size_t)(n0 + 16 + r) * K;
  const bf16* wrow2 = W + (size_t)(n0 + 32 + r) * K;
  const bf16* wrow3 = W + (size_t)(n0 + 48 + r) * K;

  // A fragment 16x32: lane r holds row; element e -> K = 16*(e>=8) + hi*8 + (e&7)
  auto ldA = [&](const bf16* row, int k0) -> v16bf {
    v8bf lo = *(const v8bf*)(row + k0 + hi * 8);
    v8bf hh = *(const v8bf*)(row + k0 + hi * 8 + 16);
    return __builtin_shufflevector(lo, hh, 0,1,2,3,4,5,6,7,8,9,10,11,12,13,14,15);
  };
  // B fragment 32x16: column n = lane&15; element e -> K = 16*(e>=8) + hi*8 + (e&7)
  auto ldB = [&](const bf16* row, int k0) -> v16bf {
    v8bf lo = *(const v8bf*)(row + k0 + hi * 8);
    v8bf hh = *(const v8bf*)(row + k0 + hi * 8 + 16);
    return __builtin_shufflevector(lo, hh, 0,1,2,3,4,5,6,7,8,9,10,11,12,13,14,15);
  };

  v8f acc[2][4] = {};

  v16bf a0 = ldA(arow0, 0), a1 = ldA(arow1, 0);
  v16bf b0 = ldB(wrow0, 0), b1 = ldB(wrow1, 0), b2 = ldB(wrow2, 0), b3 = ldB(wrow3, 0);

#define WMMA8()                                                                   \
  do {                                                                            \
    acc[0][0] = __builtin_amdgcn_wmma_f32_16x16x32_bf16(false, a0, false, b0,     \
                    (short)0, acc[0][0], false, false);                           \
    acc[1][0] = __builtin_amdgcn_wmma_f32_16x16x32_bf16(false, a1, false, b0,     \
                    (short)0, acc[1][0], false, false);                           \
    acc[0][1] = __builtin_amdgcn_wmma_f32_16x16x32_bf16(false, a0, false, b1,     \
                    (short)0, acc[0][1], false, false);                           \
    acc[1][1] = __builtin_amdgcn_wmma_f32_16x16x32_bf16(false, a1, false, b1,     \
                    (short)0, acc[1][1], false, false);                           \
    acc[0][2] = __builtin_amdgcn_wmma_f32_16x16x32_bf16(false, a0, false, b2,     \
                    (short)0, acc[0][2], false, false);                           \
    acc[1][2] = __builtin_amdgcn_wmma_f32_16x16x32_bf16(false, a1, false, b2,     \
                    (short)0, acc[1][2], false, false);                           \
    acc[0][3] = __builtin_amdgcn_wmma_f32_16x16x32_bf16(false, a0, false, b3,     \
                    (short)0, acc[0][3], false, false);                           \
    acc[1][3] = __builtin_amdgcn_wmma_f32_16x16x32_bf16(false, a1, false, b3,     \
                    (short)0, acc[1][3], false, false);                           \
  } while (0)

  for (int k0 = 0; k0 < K - 32; k0 += 32) {
    const int kn = k0 + 32;
    // issue next K-step loads first, then run WMMAs on current fragments
    v16bf na0 = ldA(arow0, kn), na1 = ldA(arow1, kn);
    v16bf nb0 = ldB(wrow0, kn), nb1 = ldB(wrow1, kn);
    v16bf nb2 = ldB(wrow2, kn), nb3 = ldB(wrow3, kn);
    __builtin_prefetch(arow0 + kn + 32, 0, 1);   // global_prefetch_b8
    WMMA8();
    a0 = na0; a1 = na1; b0 = nb0; b1 = nb1; b2 = nb2; b3 = nb3;
  }
  WMMA8();   // peeled last K-step
#undef WMMA8

#pragma unroll
  for (int mi = 0; mi < 2; ++mi) {
#pragma unroll
    for (int t = 0; t < 4; ++t) {
      const int n = n0 + t * 16 + r;
      const float bv = bias[n];
#pragma unroll
      for (int j = 0; j < 8; ++j) {
        float v = acc[mi][t][j] + bv;
        if (RELU) v = fmaxf(v, 0.f);
        // C layout: c[j] = C[hi*8+j, lane&15]
        Cout[(size_t)(m0 + mi * 16 + hi * 8 + j) * N + n] = (bf16)v;
      }
    }
  }
}

// ------------------------------------------------------------- attention
// One wave per (sequence b, head h). K,V staged in LDS; q in registers.
// MODE 0: V2E key-padding mask (CLS always valid, body from v_mask[nE,10]).
// MODE 1: E2V mask h_mask[nN,17].
template<int S, int MODE>
__global__ __launch_bounds__(128)
void attn_kernel(const bf16* __restrict__ qkv, const unsigned char* __restrict__ mask,
                 int base, bf16* __restrict__ O) {
  __shared__ __align__(16) bf16 sK[4][S * DH];
  __shared__ __align__(16) bf16 sV[4][S * DH];
  const int lane = threadIdx.x & 31;
  const int wave = threadIdx.x >> 5;
  const int p = blockIdx.x * 4 + wave;
  const int b = p >> 2;        // / NH
  const int h = p & 3;         // % NH
  const size_t rowBase = (size_t)b * S;

  for (int s = 0; s < S; ++s) {
    const bf16* kp = qkv + (rowBase + s) * (3 * D_) + D_ + h * DH;
    const bf16* vp = kp + D_;
    sK[wave][s * DH + 2 * lane]     = kp[2 * lane];
    sK[wave][s * DH + 2 * lane + 1] = kp[2 * lane + 1];
    sV[wave][s * DH + 2 * lane]     = vp[2 * lane];
    sV[wave][s * DH + 2 * lane + 1] = vp[2 * lane + 1];
  }
  __syncthreads();

  const int s1 = lane;
  if (s1 < S) {
    float qf[DH];
    const bf16* qp = qkv + (rowBase + s1) * (3 * D_) + h * DH;
#pragma unroll
    for (int i = 0; i < 8; ++i) {
      v8bf t = *(const v8bf*)(qp + i * 8);
#pragma unroll
      for (int j = 0; j < 8; ++j) qf[i * 8 + j] = (float)t[j];
    }
    float sc[S];
#pragma unroll
    for (int s2 = 0; s2 < S; ++s2) {
      bool valid;
      if (MODE == 0) valid = (s2 == 0) || (mask[(size_t)(base + b) * (SV - 1) + (s2 - 1)] != 0);
      else           valid = (mask[(size_t)(base + b) * SE + s2] != 0);
      float dacc = 0.f;
#pragma unroll
      for (int i = 0; i < 8; ++i) {
        v8bf kv = *(const v8bf*)(&sK[wave][s2 * DH + i * 8]);
#pragma unroll
        for (int j = 0; j < 8; ++j) dacc += qf[i * 8 + j] * (float)kv[j];
      }
      sc[s2] = valid ? dacc * 0.125f : -1e9f;   // 1/sqrt(64)
    }
    float mx = -1e30f;
#pragma unroll
    for (int s2 = 0; s2 < S; ++s2) mx = fmaxf(mx, sc[s2]);
    float ss = 0.f;
#pragma unroll
    for (int s2 = 0; s2 < S; ++s2) { float e = __expf(sc[s2] - mx); sc[s2] = e; ss += e; }
    const float inv = 1.f / ss;

    bf16* op = O + (rowBase + s1) * D_ + h * DH;
#pragma unroll
    for (int i = 0; i < 8; ++i) {
      float o[8] = {0.f, 0.f, 0.f, 0.f, 0.f, 0.f, 0.f, 0.f};
#pragma unroll
      for (int s2 = 0; s2 < S; ++s2) {
        v8bf vv = *(const v8bf*)(&sV[wave][s2 * DH + i * 8]);
        const float a = sc[s2] * inv;
#pragma unroll
        for (int j = 0; j < 8; ++j) o[j] += a * (float)vv[j];
      }
#pragma unroll
      for (int j = 0; j < 8; ++j) op[i * 8 + j] = (bf16)o[j];
    }
  }
}

// -------------------------------------------------- residual-add + LayerNorm
// X <- LN(X + G); also write Xb = bf16(X) for the next GEMM. One wave per row.
__global__ __launch_bounds__(256)
void add_ln_kernel(const bf16* __restrict__ G, float* __restrict__ X,
                   bf16* __restrict__ Xb, const float* __restrict__ gamma,
                   const float* __restrict__ beta, int M) {
  const int lane = threadIdx.x & 31;
  const int row = blockIdx.x * 8 + (threadIdx.x >> 5);
  if (row >= M) return;
  const size_t base = (size_t)row * D_ + lane * 8;
  float v[8];
  float s = 0.f;
#pragma unroll
  for (int i = 0; i < 8; ++i) { v[i] = X[base + i] + (float)G[base + i]; s += v[i]; }
#pragma unroll
  for (int o = 16; o > 0; o >>= 1) s += __shfl_xor(s, o, 32);
  const float mean = s * (1.f / 256.f);
  float var = 0.f;
#pragma unroll
  for (int i = 0; i < 8; ++i) { float d0 = v[i] - mean; var += d0 * d0; }
#pragma unroll
  for (int o = 16; o > 0; o >>= 1) var += __shfl_xor(var, o, 32);
  const float rstd = rsqrtf(var * (1.f / 256.f) + 1e-5f);
#pragma unroll
  for (int i = 0; i < 8; ++i) {
    const int c = lane * 8 + i;
    float y = (v[i] - mean) * rstd * gamma[c] + beta[c];
    X[base + i] = y;
    Xb[base + i] = (bf16)y;
  }
}

// ---------------------------------------------------------- gather / scatter
__global__ void gather_v2e_kernel(const float* __restrict__ unity, const float* __restrict__ pos,
                                  const int* __restrict__ xid, const int* __restrict__ xposd,
                                  int e0, float* __restrict__ X, bf16* __restrict__ Xb) {
  int idx = blockIdx.x * blockDim.x + threadIdx.x;
  if (idx >= CH * SV * D_) return;
  int d = idx & (D_ - 1);
  int s = (idx >> 8) % SV;
  int e = idx / (SV * D_);
  int ge = e0 + e;
  float v = pos[(size_t)xposd[ge * SV + s] * D_ + d];           // CLS row: zeros + pos
  if (s > 0) v += unity[(size_t)xid[ge * (SV - 1) + (s - 1)] * D_ + d];
  X[idx] = v; Xb[idx] = (bf16)v;
}

__global__ void gather_e2v_kernel(const float* __restrict__ he, const int* __restrict__ hid,
                                  int n0, float* __restrict__ X, bf16* __restrict__ Xb) {
  int idx = blockIdx.x * blockDim.x + threadIdx.x;
  if (idx >= CH * SE * D_) return;
  int d = idx & (D_ - 1);
  int s = (idx >> 8) % SE;
  int n = idx / (SE * D_);
  float v = he[(size_t)hid[(size_t)(n0 + n) * SE + s] * D_ + d];
  X[idx] = v; Xb[idx] = (bf16)v;
}

__global__ void store_cls_to_he(const float* __restrict__ X, float* __restrict__ he, int e0) {
  int idx = blockIdx.x * blockDim.x + threadIdx.x;
  if (idx >= CH * D_) return;
  int d = idx & (D_ - 1);
  int e = idx >> 8;
  he[(size_t)(e0 + e) * D_ + d] = X[(size_t)e * SV * D_ + d];
}

__global__ void scatter_cls_to_unity(const float* __restrict__ X, const int* __restrict__ subg,
                                     int n0, float* __restrict__ unity) {
  int idx = blockIdx.x * blockDim.x + threadIdx.x;
  if (idx >= CH * D_) return;
  int d = idx & (D_ - 1);
  int n = idx >> 8;
  unity[(size_t)subg[n0 + n] * D_ + d] = X[(size_t)n * SE * D_ + d];
}

__global__ void he_init_kernel(const float* __restrict__ pad_e, const float* __restrict__ cls_e,
                               float* __restrict__ he) {
  int d = threadIdx.x;
  if (d < D_) {
    he[(size_t)NE * D_ + d] = pad_e[d];
    he[(size_t)(NE + 1) * D_ + d] = cls_e[d];
  }
}

__global__ void pred_gather_kernel(const float* __restrict__ he, const int* __restrict__ pidx,
                                   float* __restrict__ out) {
  int idx = blockIdx.x * blockDim.x + threadIdx.x;
  if (idx >= BPRED * D_) return;
  int d = idx & (D_ - 1);
  int i = idx >> 8;
  out[idx] = he[(size_t)pidx[i] * D_ + d];
}

// ------------------------------------------------------------------- launch
extern "C" void kernel_launch(void* const* d_in, const int* in_sizes, int n_in,
                              void* d_out, int out_size, void* d_ws, size_t ws_size,
                              hipStream_t stream) {
  (void)in_sizes; (void)n_in; (void)out_size; (void)ws_size;
  const float* unity   = (const float*)d_in[0];
  const float* cls_emb = (const float*)d_in[1];
  const float* pad_emb = (const float*)d_in[2];
  const float* pos     = (const float*)d_in[3];
  const float* qkv_w   = (const float*)d_in[4];
  const float* qkv_b   = (const float*)d_in[5];
  const float* out_w   = (const float*)d_in[6];
  const float* out_b   = (const float*)d_in[7];
  const float* ln1_g   = (const float*)d_in[8];
  const float* ln1_b   = (const float*)d_in[9];
  const float* ln2_g   = (const float*)d_in[10];
  const float* ln2_b   = (const float*)d_in[11];
  const float* ff1_w   = (const float*)d_in[12];
  const float* ff1_b   = (const float*)d_in[13];
  const float* ff2_w   = (const float*)d_in[14];
  const float* ff2_b   = (const float*)d_in[15];
  const int* xid   = (const int*)d_in[16];
  const unsigned char* v_mask = (const unsigned char*)d_in[17];
  const int* xposd = (const int*)d_in[18];
  const int* hid   = (const int*)d_in[19];
  const unsigned char* h_mask = (const unsigned char*)d_in[20];
  const int* subg  = (const int*)d_in[21];
  const int* pidx  = (const int*)d_in[22];
  float* out = (float*)d_out;

  // ---- workspace carve (≈350 MB)
  char* wsc = (char*)d_ws;
  size_t off = 0;
  auto carve = [&](size_t bytes) -> void* {
    void* p = wsc + off;
    off += (bytes + 255) & ~(size_t)255;
    return p;
  };
  bf16*  wqkv16  = (bf16*)carve(sizeof(bf16) * 2 * 768 * 256);
  bf16*  wo16    = (bf16*)carve(sizeof(bf16) * 2 * 256 * 256);
  bf16*  w116    = (bf16*)carve(sizeof(bf16) * 2 * 1024 * 256);
  bf16*  w216    = (bf16*)carve(sizeof(bf16) * 2 * 256 * 1024);
  float* unity_w = (float*)carve(sizeof(float) * (size_t)NTOT * D_);
  float* he      = (float*)carve(sizeof(float) * (size_t)(NE + 2) * D_);
  float* X       = (float*)carve(sizeof(float) * (size_t)CH * SE * D_);
  bf16*  Xb      = (bf16*)carve(sizeof(bf16) * (size_t)CH * SE * D_);
  bf16*  QKV     = (bf16*)carve(sizeof(bf16) * (size_t)CH * SE * 3 * D_);
  bf16*  Oa      = (bf16*)carve(sizeof(bf16) * (size_t)CH * SE * D_);
  bf16*  Gb      = (bf16*)carve(sizeof(bf16) * (size_t)CH * SE * D_);
  bf16*  Fb      = (bf16*)carve(sizeof(bf16) * (size_t)CH * SE * FF_);

  // ---- one-time per call: bf16 weights, working unity copy, he pad/cls rows
  auto cvt = [&](const float* s, bf16* d, int n) {
    f2bf_kernel<<<(n + 255) / 256, 256, 0, stream>>>(s, d, n);
  };
  cvt(qkv_w, wqkv16, 2 * 768 * 256);
  cvt(out_w, wo16,   2 * 256 * 256);
  cvt(ff1_w, w116,   2 * 1024 * 256);
  cvt(ff2_w, w216,   2 * 256 * 1024);
  hipMemcpyAsync(unity_w, unity, sizeof(float) * (size_t)NTOT * D_,
                 hipMemcpyDeviceToDevice, stream);
  he_init_kernel<<<1, 256, 0, stream>>>(pad_emb, cls_emb, he);

  // ---- one transformer layer on the current chunk
  auto run_layer = [&](int S, int M, int l, int mode, int base,
                       const unsigned char* mask) {
    const bf16* Wq = wqkv16 + (size_t)l * 768 * 256;
    const bf16* Wo = wo16   + (size_t)l * 256 * 256;
    const bf16* W1 = w116   + (size_t)l * 1024 * 256;
    const bf16* W2 = w216   + (size_t)l * 256 * 1024;
    gemm_bf16_wmma<false><<<dim3(768 / 128, M / 128), 256, 0, stream>>>(
        Xb, Wq, qkv_b + (size_t)l * 768, QKV, M, 768, 256);
    if (mode == 0)
      attn_kernel<SV, 0><<<CH * NH / 4, 128, 0, stream>>>(QKV, mask, base, Oa);
    else
      attn_kernel<SE, 1><<<CH * NH / 4, 128, 0, stream>>>(QKV, mask, base, Oa);
    gemm_bf16_wmma<false><<<dim3(256 / 128, M / 128), 256, 0, stream>>>(
        Oa, Wo, out_b + (size_t)l * 256, Gb, M, 256, 256);
    add_ln_kernel<<<M / 8, 256, 0, stream>>>(
        Gb, X, Xb, ln1_g + (size_t)l * 256, ln1_b + (size_t)l * 256, M);
    gemm_bf16_wmma<true><<<dim3(1024 / 128, M / 128), 256, 0, stream>>>(
        Xb, W1, ff1_b + (size_t)l * 1024, Fb, M, 1024, 256);
    gemm_bf16_wmma<false><<<dim3(256 / 128, M / 128), 256, 0, stream>>>(
        Fb, W2, ff2_b + (size_t)l * 256, Gb, M, 256, 1024);
    add_ln_kernel<<<M / 8, 256, 0, stream>>>(
        Gb, X, Xb, ln2_g + (size_t)l * 256, ln2_b + (size_t)l * 256, M);
  };

  const int MV = CH * SV;   // 22528 (÷128)
  const int ME = CH * SE;   // 34816 (÷128)

  for (int k = 0; k <= KHOP; ++k) {
    // V2E: hyperedge encoder, write CLS rows into he table
    for (int c = 0; c < NE / CH; ++c) {
      const int e0 = c * CH;
      gather_v2e_kernel<<<(CH * SV * D_) / 256, 256, 0, stream>>>(
          unity_w, pos, xid, xposd, e0, X, Xb);
      for (int l = 0; l < 2; ++l) run_layer(SV, MV, l, 0, e0, v_mask);
      store_cls_to_he<<<(CH * D_) / 256, 256, 0, stream>>>(X, he, e0);
    }
    if (k == KHOP) break;
    // E2V: node encoder over he table (+pad/cls rows), scatter CLS into unity
    for (int c = 0; c < NN / CH; ++c) {
      const int n0 = c * CH;
      gather_e2v_kernel<<<(CH * SE * D_) / 256, 256, 0, stream>>>(
          he, hid, n0, X, Xb);
      for (int l = 0; l < 2; ++l) run_layer(SE, ME, l, 1, n0, h_mask);
      scatter_cls_to_unity<<<(CH * D_) / 256, 256, 0, stream>>>(X, subg, n0, unity_w);
    }
  }
  pred_gather_kernel<<<(BPRED * D_) / 256, 256, 0, stream>>>(he, pidx, out);
}
